// ResidualQuantizer_80367428043180
// MI455X (gfx1250) — matrix operand order repaired and checked by reading, not compiled, not run on hardware
//
#include <hip/hip_runtime.h>
#include <math.h>

typedef __bf16 bf16;
typedef bf16  bf16x4  __attribute__((ext_vector_type(4)));
typedef bf16  bf16x8  __attribute__((ext_vector_type(8)));
typedef bf16  bf16x16 __attribute__((ext_vector_type(16)));
typedef float floatx8 __attribute__((ext_vector_type(8)));

#define B_  8
#define S_  2048
#define D_  512
#define Q_  8
#define K_  1024
#define N_  (B_ * S_)            // 16384 tokens
#define ND_ ((size_t)N_ * D_)    // 8388608
#define AROW 520                 // 512 + 8 bf16 padding -> conflict-free LDS rows
#define MTOK 128                 // tokens staged per block (4 waves x 32 tokens)

// ---------------- utility kernels ----------------

__global__ void zero_kernel(float* p, int n) {
    for (int i = blockIdx.x * blockDim.x + threadIdx.x; i < n; i += gridDim.x * blockDim.x)
        p[i] = 0.0f;
}

__global__ void copy4_kernel(float4* dst, const float4* src, int n4) {
    for (int i = blockIdx.x * blockDim.x + threadIdx.x; i < n4; i += gridDim.x * blockDim.x)
        dst[i] = src[i];
}

// split fp32 codebooks into bf16 hi/lo pair (bf16x3 emulation of fp32 dot)
__global__ void cvt_codebook(const float* cb, bf16* chi, bf16* clo, int n) {
    for (int i = blockIdx.x * blockDim.x + threadIdx.x; i < n; i += gridDim.x * blockDim.x) {
        float f = cb[i];
        bf16 h = (bf16)f;
        chi[i] = h;
        clo[i] = (bf16)(f - (float)h);
    }
}

// ||c||^2 per (level, code): one wave per row
__global__ void cnorm_kernel(const float* cb, float* cnorm) {
    int row  = blockIdx.x;            // Q_*K_ rows
    int lane = threadIdx.x;           // 32
    const float* c = cb + (size_t)row * D_;
    float s = 0.0f;
    for (int i = lane; i < D_; i += 32) { float v = c[i]; s += v * v; }
    for (int off = 16; off > 0; off >>= 1) s += __shfl_down(s, off, 32);
    if (lane == 0) cnorm[row] = s;
}

// ---------------- distance + argmin (WMMA bf16x3, 2-way M-blocked) ----------------
// block = 128 threads (4 waves); each wave owns 32 tokens (two 16-row A tiles that
// share every B load -> halves codebook L2 traffic). grid = N_/128 per level.
__global__ void __launch_bounds__(128)
dist_kernel(const float* __restrict__ residual,
            const bf16* __restrict__ chi, const bf16* __restrict__ clo,
            const float* __restrict__ cnorm, int* __restrict__ idxbuf, int q) {
    extern __shared__ bf16 smem[];
    bf16* ahi = smem;                 // [MTOK][AROW]
    bf16* alo = smem + MTOK * AROW;

    const int tid     = threadIdx.x;
    const int tokbase = blockIdx.x * MTOK;

    // stage MTOK tokens x 512 dims of residual, split to bf16 hi/lo, into LDS
    const float4* rsrc = (const float4*)(residual + (size_t)tokbase * D_);
    for (int i = tid; i < MTOK * (D_ / 4); i += 128) {
        int tok  = i >> 7;            // / (D_/4)
        int dp   = (i & 127) << 2;
        float4 v = rsrc[i];
        bf16 h0 = (bf16)v.x, h1 = (bf16)v.y, h2 = (bf16)v.z, h3 = (bf16)v.w;
        bf16x4 hv = {h0, h1, h2, h3};
        bf16x4 lv = {(bf16)(v.x - (float)h0), (bf16)(v.y - (float)h1),
                     (bf16)(v.z - (float)h2), (bf16)(v.w - (float)h3)};
        *(bf16x4*)(ahi + tok * AROW + dp) = hv;
        *(bf16x4*)(alo + tok * AROW + dp) = lv;
    }
    __syncthreads();

    const int lane = tid & 31;
    const int wave = tid >> 5;        // 4 waves, 32 tokens each
    const int col  = lane & 15;       // B/C column slot (code within tile)
    const int half = lane >> 4;       // K sub-block select (K 0..15 / 16..31 for A and B)

    float best0[8], best1[8];
    int   bidx0[8], bidx1[8];
#pragma unroll
    for (int r = 0; r < 8; r++) {
        best0[r] = 3.4e38f; bidx0[r] = 0;
        best1[r] = 3.4e38f; bidx1[r] = 0;
    }

    const size_t cboff = (size_t)q * (size_t)K_ * D_;
    const bf16* Chi = chi + cboff;
    const bf16* Clo = clo + cboff;
    // A rows: tile0 = wave*32 + col, tile1 = wave*32 + 16 + col
    const bf16* a0h = ahi + (wave * 32 + col) * AROW + half * 16;
    const bf16* a0l = alo + (wave * 32 + col) * AROW + half * 16;
    const bf16* a1h = a0h + 16 * AROW;
    const bf16* a1l = a0l + 16 * AROW;
    const float* cnq = cnorm + q * K_;

    for (int tile = 0; tile < 64; ++tile) {
        const int code = tile * 16 + col;
        floatx8 acc0 = {0.f, 0.f, 0.f, 0.f, 0.f, 0.f, 0.f, 0.f};
        floatx8 acc1 = {0.f, 0.f, 0.f, 0.f, 0.f, 0.f, 0.f, 0.f};
        // B: lane holds 16 contiguous K values (half selects K 0..15 / 16..31)
        const bf16* bhp = Chi + (size_t)code * D_ + half * 16;
        const bf16* blp = Clo + (size_t)code * D_ + half * 16;
#pragma unroll 4
        for (int c = 0; c < 16; ++c) {           // D in chunks of 32
            bf16x16 Bhi = *(const bf16x16*)(bhp + c * 32);
            bf16x16 Blo = *(const bf16x16*)(blp + c * 32);
            bf16x8 t0, t1;
            t0 = *(const bf16x8*)(a0h + c * 32);
            t1 = *(const bf16x8*)(a0h + c * 32 + 8);
            bf16x16 A0h = __builtin_shufflevector(t0, t1, 0,1,2,3,4,5,6,7,8,9,10,11,12,13,14,15);
            t0 = *(const bf16x8*)(a0l + c * 32);
            t1 = *(const bf16x8*)(a0l + c * 32 + 8);
            bf16x16 A0l = __builtin_shufflevector(t0, t1, 0,1,2,3,4,5,6,7,8,9,10,11,12,13,14,15);
            t0 = *(const bf16x8*)(a1h + c * 32);
            t1 = *(const bf16x8*)(a1h + c * 32 + 8);
            bf16x16 A1h = __builtin_shufflevector(t0, t1, 0,1,2,3,4,5,6,7,8,9,10,11,12,13,14,15);
            t0 = *(const bf16x8*)(a1l + c * 32);
            t1 = *(const bf16x8*)(a1l + c * 32 + 8);
            bf16x16 A1l = __builtin_shufflevector(t0, t1, 0,1,2,3,4,5,6,7,8,9,10,11,12,13,14,15);
            acc0 = __builtin_amdgcn_wmma_f32_16x16x32_bf16(false, A0h, false, Bhi, (short)0, acc0, false, false);
            acc1 = __builtin_amdgcn_wmma_f32_16x16x32_bf16(false, A1h, false, Bhi, (short)0, acc1, false, false);
            acc0 = __builtin_amdgcn_wmma_f32_16x16x32_bf16(false, A0h, false, Blo, (short)0, acc0, false, false);
            acc1 = __builtin_amdgcn_wmma_f32_16x16x32_bf16(false, A1h, false, Blo, (short)0, acc1, false, false);
            acc0 = __builtin_amdgcn_wmma_f32_16x16x32_bf16(false, A0l, false, Bhi, (short)0, acc0, false, false);
            acc1 = __builtin_amdgcn_wmma_f32_16x16x32_bf16(false, A1l, false, Bhi, (short)0, acc1, false, false);
        }
        const float cn = cnq[code];
#pragma unroll
        for (int r = 0; r < 8; r++) {            // dist = ||c||^2 - 2 r.c (argmin-equivalent)
            float d0 = cn - 2.0f * acc0[r];
            if (d0 < best0[r]) { best0[r] = d0; bidx0[r] = code; }
            float d1 = cn - 2.0f * acc1[r];
            if (d1 < best1[r]) { best1[r] = d1; bidx1[r] = code; }
        }
    }

    // min+index reduction across the 16 column lanes (xor masks stay in-half)
#pragma unroll
    for (int s = 1; s < 16; s <<= 1) {
#pragma unroll
        for (int r = 0; r < 8; r++) {
            float ob = __shfl_xor(best0[r], s, 32);
            int   oi = __shfl_xor(bidx0[r], s, 32);
            if (ob < best0[r] || (ob == best0[r] && oi < bidx0[r])) { best0[r] = ob; bidx0[r] = oi; }
            ob = __shfl_xor(best1[r], s, 32);
            oi = __shfl_xor(bidx1[r], s, 32);
            if (ob < best1[r] || (ob == best1[r] && oi < bidx1[r])) { best1[r] = ob; bidx1[r] = oi; }
        }
    }
    if (col == 0) {                   // lane 0 -> C rows 0..7, lane 16 -> rows 8..15
        int tok0 = tokbase + wave * 32 + half * 8;
#pragma unroll
        for (int r = 0; r < 8; r++) {
            idxbuf[tok0 + r]      = bidx0[r];
            idxbuf[tok0 + 16 + r] = bidx1[r];
        }
    }
}

// ---------------- gather + residual/qsum/commit/histogram ----------------
__global__ void __launch_bounds__(128)
update_kernel(float* __restrict__ residual, const float* __restrict__ codebooks,
              const int* __restrict__ idxbuf, float* __restrict__ qsum,
              float* __restrict__ idx_out, float* __restrict__ counts,
              float* __restrict__ commit, int q) {
    const int token = blockIdx.x;
    const int tid   = threadIdx.x;    // 128, 4 floats each
    const int idx   = idxbuf[token];
    const float4* cb = (const float4*)(codebooks + ((size_t)q * K_ + idx) * D_);
    float4* r4 = (float4*)(residual + (size_t)token * D_);
    float4* s4 = (float4*)(qsum + (size_t)token * D_);
    float4 r = r4[tid], c = cb[tid];
    float4 d = {r.x - c.x, r.y - c.y, r.z - c.z, r.w - c.w};
    r4[tid] = d;
    float4 s = s4[tid];
    s.x += c.x; s.y += c.y; s.z += c.z; s.w += c.w;
    s4[tid] = s;
    float loc = d.x * d.x + d.y * d.y + d.z * d.z + d.w * d.w;
    for (int off = 16; off > 0; off >>= 1) loc += __shfl_down(loc, off, 32);
    __shared__ float part[4];
    if ((tid & 31) == 0) part[tid >> 5] = loc;
    __syncthreads();
    if (tid == 0) {
        atomicAdd(&commit[q], part[0] + part[1] + part[2] + part[3]);
        atomicAdd(&counts[q * K_ + idx], 1.0f);
        idx_out[(size_t)token * Q_ + q] = (float)idx;   // (B,S,Q) token-major
    }
}

// ---------------- scalars: commitment loss + perplexity ----------------
__global__ void finalize_kernel(const float* counts, const float* commit, float* out2) {
    __shared__ float red[1024];
    const int t = threadIdx.x;        // 1024 == K_
    float perp_sum = 0.0f;
    for (int q = 0; q < Q_; q++) {
        float p = counts[q * K_ + t] / (float)N_;
        red[t] = p * __logf(p + 1e-10f);
        __syncthreads();
        for (int s = 512; s > 0; s >>= 1) {
            if (t < s) red[t] += red[t + s];
            __syncthreads();
        }
        if (t == 0) perp_sum += __expf(-red[0]);
        __syncthreads();
    }
    if (t == 0) {
        float cl = 0.0f;
        for (int q = 0; q < Q_; q++) cl += commit[q];
        out2[0] = 0.25f * cl / (float)ND_;  // commitment loss
        out2[1] = perp_sum / (float)Q_;     // perplexity
    }
}

// ---------------- host orchestration ----------------
extern "C" void kernel_launch(void* const* d_in, const int* in_sizes, int n_in,
                              void* d_out, int out_size, void* d_ws, size_t ws_size,
                              hipStream_t stream) {
    (void)in_sizes; (void)n_in; (void)out_size; (void)ws_size;
    const float* x  = (const float*)d_in[0];
    const float* cb = (const float*)d_in[1];

    float* out     = (float*)d_out;
    float* qsum    = out;                       // quantized == qsum (straight-through)
    float* idx_out = out + ND_;                 // N*Q floats
    float* scal    = idx_out + (size_t)N_ * Q_; // [loss, perplexity]

    // workspace layout
    float* residual = (float*)d_ws;                         // N*D fp32   (32 MB)
    bf16*  chi      = (bf16*)(residual + ND_);              // Q*K*D bf16 ( 8 MB)
    bf16*  clo      = chi + (size_t)Q_ * K_ * D_;           // Q*K*D bf16 ( 8 MB)
    float* cnorm    = (float*)(clo + (size_t)Q_ * K_ * D_); // Q*K
    float* counts   = cnorm + Q_ * K_;                      // Q*K
    float* commit   = counts + Q_ * K_;                     // Q
    int*   idxbuf   = (int*)(commit + Q_ + 8);              // N ints

    zero_kernel<<<1024, 256, 0, stream>>>(qsum, (int)ND_);
    zero_kernel<<<8, 256, 0, stream>>>(counts, Q_ * K_ + Q_);  // counts+commit contiguous
    copy4_kernel<<<1024, 256, 0, stream>>>((float4*)residual, (const float4*)x, (int)(ND_ / 4));
    cvt_codebook<<<2048, 256, 0, stream>>>(cb, chi, clo, Q_ * K_ * D_);
    cnorm_kernel<<<Q_ * K_, 32, 0, stream>>>(cb, cnorm);

    const size_t lds_bytes = (size_t)2 * MTOK * AROW * sizeof(bf16); // 266,240 B
    for (int q = 0; q < Q_; q++) {
        dist_kernel<<<N_ / MTOK, 128, lds_bytes, stream>>>(residual, chi, clo, cnorm, idxbuf, q);
        update_kernel<<<N_, 128, 0, stream>>>(residual, cb, idxbuf, qsum, idx_out, counts, commit, q);
    }
    finalize_kernel<<<1, 1024, 0, stream>>>(counts, commit, scal);
}
